// MultiHeadAttention_72971494359529
// MI455X (gfx1250) — matrix operand order repaired and checked by reading, not compiled, not run on hardware
//
#include <hip/hip_runtime.h>

// ---------------------------------------------------------------------------
// CDNA5 (gfx1250) wave32 WMMA attention pipeline, all matmuls on
// v_wmma_f32_16x16x32_bf16 (fp32 -> bf16 RNE on the fly, f32 accumulate).
// K/V tiles in the attention kernel are fetched by the Tensor Data Mover
// (tensor_load_to_lds + s_wait_tensorcnt).
// ---------------------------------------------------------------------------

typedef __attribute__((ext_vector_type(16))) __bf16    v16bf;
typedef __attribute__((ext_vector_type(8)))  float     v8f;
typedef __attribute__((ext_vector_type(4)))  unsigned  u32x4;
typedef __attribute__((ext_vector_type(8)))  unsigned  u32x8;

constexpr int B_    = 2;
constexpr int T_    = 2048;
constexpr int C_    = 2048;
constexpr int H_    = 32;
constexpr int KVH_  = 8;
constexpr int HD_   = 64;
constexpr int NREP_ = H_ / KVH_;

#define DEVFN __device__ __forceinline__

DEVFN __bf16 f2bf(float x) {
  unsigned u = __float_as_uint(x);
  u += 0x7FFFu + ((u >> 16) & 1u);          // round-to-nearest-even
  union { unsigned short s; __bf16 b; } cv;
  cv.s = (unsigned short)(u >> 16);
  return cv.b;
}
DEVFN float bf2f(__bf16 x) {
  union { unsigned short s; __bf16 b; } cv;
  cv.b = x;
  return __uint_as_float((unsigned)cv.s << 16);
}
DEVFN __bf16 to_bf(float x)  { return f2bf(x); }
DEVFN __bf16 to_bf(__bf16 x) { return x; }
DEVFN void stc(float* p,  float v) { *p = v; }
DEVFN void stc(__bf16* p, float v) { *p = f2bf(v); }

DEVFN v8f wmma_bf16(v16bf a, v16bf b, v8f c) {
  // (neg_a, A, neg_b, B, c_mod, C, reuse_a, reuse_b)
  return __builtin_amdgcn_wmma_f32_16x16x32_bf16(false, a, false, b,
                                                 (short)0, c, false, false);
}

// ---------------------------------------------------------------------------
// TDM: issue a 2D tensor_load_to_lds.  rows x cols bf16 tile, source row
// pitch row_stride_elems (bf16 units), LDS dest padded with 4 DWORDs after
// every 32 DWORDs (128B row -> 144B LDS pitch, matching [64][72] tiles).
// Wave-level op (EXEC ignored); caller gates to one wave per workgroup.
// ---------------------------------------------------------------------------
DEVFN void tdm_load_tile_bf16(unsigned lds_off, const __bf16* gptr,
                              unsigned rows, unsigned cols,
                              unsigned row_stride_elems) {
  unsigned long long ga = (unsigned long long)(size_t)gptr;
  u32x4 g0;
  g0[0] = 1u;                                          // count=1 (valid user D#)
  g0[1] = lds_off;                                     // lds_addr [63:32]
  g0[2] = (unsigned)(ga & 0xFFFFFFFFu);                // global_addr [95:64]
  g0[3] = (unsigned)((ga >> 32) & 0x01FFFFFFu)         // global_addr [120:96]
        | (2u << 30);                                  // type=2 ("image")
  u32x8 g1;
  g1[0] = (1u << 16)                                   // data_size = 2 bytes
        | (1u << 20)                                   // pad_enable
        | (4u << 22)                                   // pad_interval: 32 DWORDs
        | (3u << 25);                                  // pad_amount: 4 DWORDs
  g1[1] = (cols & 0xFFFFu) << 16;                      // tensor_dim0[15:0]
  g1[2] = (rows & 0xFFFFu) << 16;                      // tensor_dim1[15:0]
  g1[3] = (cols & 0xFFFFu) << 16;                      // tile_dim0
  g1[4] = rows & 0xFFFFu;                              // tile_dim1 (tile_dim2=0)
  g1[5] = row_stride_elems;                            // tensor_dim0_stride[31:0]
  g1[6] = 0u;
  g1[7] = 0u;
  asm volatile("tensor_load_to_lds %0, %1" :: "s"(g0), "s"(g1) : "memory");
}

// ---------------------------------------------------------------------------
// Tiled GEMM: C[M,N] = A[M,K] @ B[K,N].  BM=128 BN=128 BK=32, 8 waves,
// each wave computes a 32x64 sub-tile = 2x4 WMMA tiles.
// ---------------------------------------------------------------------------
template <typename TA, typename TB, typename TC>
__global__ __launch_bounds__(256) void gemm_wmma_kernel(
    const TA* __restrict__ A, const TB* __restrict__ Bm, TC* __restrict__ Cm,
    int M, int N, int K) {
  (void)M;
  constexpr int BM = 128, BN = 128, BK = 32;
  __shared__ __bf16 As[BM][BK + 8];
  __shared__ __bf16 Bs[BK][BN + 8];

  const int tid  = threadIdx.x;
  const int lane = tid & 31;
  const int wave = tid >> 5;
  const int l    = lane & 15;
  const int hi   = lane >> 4;
  const int wm   = (wave & 3) * 32;
  const int wn   = (wave >> 2) * 64;
  const int bm0  = blockIdx.y * BM;
  const int bn0  = blockIdx.x * BN;

  const v8f zero8 = {0.f, 0.f, 0.f, 0.f, 0.f, 0.f, 0.f, 0.f};
  v8f acc[2][4];
#pragma unroll
  for (int i = 0; i < 2; ++i)
#pragma unroll
    for (int j = 0; j < 4; ++j) acc[i][j] = zero8;

  const int ar = tid >> 1, ac = (tid & 1) * 16;   // A tile: 128 rows x 32
  const int br = tid >> 3, bc = (tid & 7) * 16;   // B tile: 32 rows x 128

  for (int k0 = 0; k0 < K; k0 += BK) {
    const TA* asrc = A  + (size_t)(bm0 + ar) * K + (k0 + ac);
    const TB* bsrc = Bm + (size_t)(k0 + br) * N + (bn0 + bc);
#pragma unroll
    for (int i = 0; i < 16; ++i) As[ar][ac + i] = to_bf(asrc[i]);
#pragma unroll
    for (int i = 0; i < 16; ++i) Bs[br][bc + i] = to_bf(bsrc[i]);
    if (k0 + BK < K) {                        // global_prefetch_b8 next tiles
      __builtin_prefetch(asrc + BK, 0, 3);
      __builtin_prefetch(bsrc + (size_t)BK * N, 0, 3);
    }
    __syncthreads();

    v16bf fa[2], fb[4];
#pragma unroll
    for (int i = 0; i < 2; ++i) {             // A frag: 16x32, k = 16*hi + e (contiguous)
      const int row = wm + i * 16 + l;
#pragma unroll
      for (int e = 0; e < 16; ++e)
        fa[i][e] = As[row][e + 16 * hi];
    }
#pragma unroll
    for (int j = 0; j < 4; ++j) {             // B frag: 32x16, k = 16*hi + e (contiguous)
      const int col = wn + j * 16 + l;
#pragma unroll
      for (int e = 0; e < 16; ++e)
        fb[j][e] = Bs[e + 16 * hi][col];
    }
#pragma unroll
    for (int i = 0; i < 2; ++i)
#pragma unroll
      for (int j = 0; j < 4; ++j)
        acc[i][j] = wmma_bf16(fa[i], fb[j], acc[i][j]);
    __syncthreads();
  }

#pragma unroll
  for (int i = 0; i < 2; ++i)
#pragma unroll
    for (int j = 0; j < 4; ++j) {
      const int col = bn0 + wn + j * 16 + l;
#pragma unroll
      for (int r = 0; r < 8; ++r) {           // C/D layout: M = r + 8*hi
        const int row = bm0 + wm + i * 16 + r + 8 * hi;
        stc(&Cm[(size_t)row * N + col], acc[i][j][r]);
      }
    }
}

// ---------------------------------------------------------------------------
// RoPE, in place on bf16 q/k stored (B, T, nh, HD). One thread per pair.
// ---------------------------------------------------------------------------
__global__ __launch_bounds__(256) void rope_kernel(__bf16* t,
                                                   const float* __restrict__ freq,
                                                   int nh, size_t total) {
  size_t idx = (size_t)blockIdx.x * blockDim.x + threadIdx.x;
  if (idx >= total) return;
  const int p = (int)(idx & 31);
  size_t rest = idx >> 5;
  const int hh = (int)(rest % nh); rest /= (size_t)nh;
  const int tt = (int)(rest % T_);
  const size_t bb = rest / T_;
  const float c = freq[((size_t)tt * 32 + p) * 2 + 0];
  const float s = freq[((size_t)tt * 32 + p) * 2 + 1];
  __bf16* base = t + (((bb * T_ + tt) * (size_t)nh + hh) * HD_ + 2 * p);
  const float x0 = bf2f(base[0]);
  const float x1 = bf2f(base[1]);
  base[0] = f2bf(x0 * c - x1 * s);
  base[1] = f2bf(x1 * c + x0 * s);
}

// ---------------------------------------------------------------------------
// Flash attention: block = (b,h, q-tile of 64 rows), 4 waves x 16 q rows.
// q (B,T,H,HD) bf16, k/v (B,T,KVH,HD) bf16, o (B,T,H,HD) bf16.
// K/V tiles are DMA'd into LDS by the Tensor Data Mover.
// ---------------------------------------------------------------------------
__global__ __launch_bounds__(128) void attn_kernel(const __bf16* __restrict__ q,
                                                   const __bf16* __restrict__ k,
                                                   const __bf16* __restrict__ v,
                                                   __bf16* __restrict__ o) {
  const int bh  = blockIdx.x;
  const int b   = bh / H_;
  const int h   = bh % H_;
  const int kvh = h / NREP_;
  const int q0  = blockIdx.y * 64;

  __shared__ __bf16 Ks[64][HD_ + 8];   // 144B LDS row pitch (TDM pad fields)
  __shared__ __bf16 Vs[64][HD_ + 8];
  __shared__ __bf16 Ps[4][16][64 + 8];

  const int tid  = threadIdx.x;
  const int lane = tid & 31;
  const int wave = tid >> 5;
  const int l    = lane & 15;
  const int hi   = lane >> 4;
  const int wq   = wave * 16;

  const unsigned ks_off = (unsigned)(size_t)(&Ks[0][0]);  // flat->LDS offset
  const unsigned vs_off = (unsigned)(size_t)(&Vs[0][0]);

  // Q fragments (A layout): rows q0+wq..+15, HD=64 as two K=32 chunks.
  v16bf qf[2];
  {
    const int qi = q0 + wq + l;
    const __bf16* qrow = q + (((size_t)b * T_ + qi) * H_ + h) * HD_;
#pragma unroll
    for (int c = 0; c < 2; ++c)
#pragma unroll
      for (int e = 0; e < 16; ++e)
        qf[c][e] = qrow[c * 32 + e + 16 * hi];
  }

  const v8f zero8 = {0.f, 0.f, 0.f, 0.f, 0.f, 0.f, 0.f, 0.f};
  float mrow[8], lrow[8];
  v8f oacc[4];
#pragma unroll
  for (int r = 0; r < 8; ++r) { mrow[r] = -1e30f; lrow[r] = 0.f; }
#pragma unroll
  for (int j = 0; j < 4; ++j) oacc[j] = zero8;

  const int ktiles = q0 / 64 + 1;               // causal: only tiles <= q tile
  for (int it = 0; it < ktiles; ++it) {
    const int kt0 = it * 64;

    // Tensor Data Mover fetch of K/V 64x64 bf16 tiles (row pitch 512 elems).
    if (wave == 0) {
      const __bf16* ksrc = k + (((size_t)b * T_ + kt0) * KVH_ + kvh) * HD_;
      const __bf16* vsrc = v + (((size_t)b * T_ + kt0) * KVH_ + kvh) * HD_;
      tdm_load_tile_bf16(ks_off, ksrc, 64u, 64u, (unsigned)(KVH_ * HD_));
      tdm_load_tile_bf16(vs_off, vsrc, 64u, 64u, (unsigned)(KVH_ * HD_));
    }
    __builtin_amdgcn_s_wait_tensorcnt(0);
    __syncthreads();

    // S = Q @ K^T : B frag rows are the d-axis -> contiguous reads of Ks rows.
    v8f sacc[4];
#pragma unroll
    for (int j = 0; j < 4; ++j) sacc[j] = zero8;
#pragma unroll
    for (int j = 0; j < 4; ++j) {
      v16bf fb0, fb1;
      const int kc = j * 16 + l;                // column of S = key index
#pragma unroll
      for (int e = 0; e < 16; ++e) {
        fb0[e] = Ks[kc][e + 16 * hi];
        fb1[e] = Ks[kc][32 + e + 16 * hi];
      }
      sacc[j] = wmma_bf16(qf[0], fb0, sacc[j]);
      sacc[j] = wmma_bf16(qf[1], fb1, sacc[j]);
    }

    // Online softmax; rows r+8*hi live in 16-lane halves -> xor<16 reductions.
#pragma unroll
    for (int r = 0; r < 8; ++r) {
      const int row = q0 + wq + r + 8 * hi;
      float mx = -1e30f;
#pragma unroll
      for (int j = 0; j < 4; ++j) {
        const int col = kt0 + j * 16 + l;
        float sv = sacc[j][r] * 0.125f + (col > row ? -1.0e9f : 0.0f);
        sacc[j][r] = sv;
        mx = fmaxf(mx, sv);
      }
#pragma unroll
      for (int d = 1; d < 16; d <<= 1) mx = fmaxf(mx, __shfl_xor(mx, d, 32));
      const float mnew  = fmaxf(mrow[r], mx);
      const float alpha = __expf(mrow[r] - mnew);
      float rs = 0.f;
#pragma unroll
      for (int j = 0; j < 4; ++j) {
        const float pv = __expf(sacc[j][r] - mnew);
        Ps[wave][r + 8 * hi][j * 16 + l] = f2bf(pv);   // C-layout -> LDS
        rs += pv;
      }
#pragma unroll
      for (int d = 1; d < 16; d <<= 1) rs += __shfl_xor(rs, d, 32);
      lrow[r] = lrow[r] * alpha + rs;
      mrow[r] = mnew;
#pragma unroll
      for (int jd = 0; jd < 4; ++jd) oacc[jd][r] *= alpha;
    }
    // same-wave LDS store->load ordering for the P reshape
    asm volatile("s_wait_dscnt 0x0" ::: "memory");

    // O += P @ V : reload P in A-layout from this wave's private Ps region.
#pragma unroll
    for (int c = 0; c < 2; ++c) {
      v16bf fp;
#pragma unroll
      for (int e = 0; e < 16; ++e)
        fp[e] = Ps[wave][l][c * 32 + e + 16 * hi];
#pragma unroll
      for (int jd = 0; jd < 4; ++jd) {
        v16bf fv;
#pragma unroll
        for (int e = 0; e < 16; ++e)
          fv[e] = Vs[c * 32 + e + 16 * hi][jd * 16 + l];
        oacc[jd] = wmma_bf16(fp, fv, oacc[jd]);
      }
    }
    __syncthreads();
  }

#pragma unroll
  for (int jd = 0; jd < 4; ++jd) {
    const int col = jd * 16 + l;
#pragma unroll
    for (int r = 0; r < 8; ++r) {
      const int row = q0 + wq + r + 8 * hi;
      const float ov = oacc[jd][r] / lrow[r];
      o[(((size_t)b * T_ + row) * H_ + h) * HD_ + col] = f2bf(ov);
    }
  }
}

// ---------------------------------------------------------------------------
// Host launcher. Inputs: x, freq_cis, mask(unused), window(unused), Wq,Wk,Wv,Wo
// ---------------------------------------------------------------------------
extern "C" void kernel_launch(void* const* d_in, const int* in_sizes, int n_in,
                              void* d_out, int out_size, void* d_ws, size_t ws_size,
                              hipStream_t stream) {
  (void)in_sizes; (void)n_in; (void)out_size; (void)ws_size;
  const float* x    = (const float*)d_in[0];
  const float* freq = (const float*)d_in[1];
  const float* Wq   = (const float*)d_in[4];
  const float* Wk   = (const float*)d_in[5];
  const float* Wv   = (const float*)d_in[6];
  const float* Wo   = (const float*)d_in[7];

  const size_t MT = (size_t)B_ * T_;            // 4096 rows
  __bf16* qb = (__bf16*)d_ws;                   // (B,T,H,HD)   bf16
  __bf16* kb = qb + MT * (size_t)(H_ * HD_);    // (B,T,KVH,HD) bf16
  __bf16* vb = kb + MT * (size_t)(KVH_ * HD_);  // (B,T,KVH,HD) bf16
  __bf16* ab = vb + MT * (size_t)(KVH_ * HD_);  // (B,T,H,HD)   bf16

  const dim3 blk(256);
  gemm_wmma_kernel<float, float, __bf16>
      <<<dim3((H_ * HD_) / 128, (int)(MT / 128)), blk, 0, stream>>>(
          x, Wq, qb, (int)MT, H_ * HD_, C_);
  gemm_wmma_kernel<float, float, __bf16>
      <<<dim3((KVH_ * HD_) / 128, (int)(MT / 128)), blk, 0, stream>>>(
          x, Wk, kb, (int)MT, KVH_ * HD_, C_);
  gemm_wmma_kernel<float, float, __bf16>
      <<<dim3((KVH_ * HD_) / 128, (int)(MT / 128)), blk, 0, stream>>>(
          x, Wv, vb, (int)MT, KVH_ * HD_, C_);

  {
    const size_t tq = MT * (size_t)H_ * 32;
    rope_kernel<<<(unsigned)((tq + 255) / 256), 256, 0, stream>>>(qb, freq, H_, tq);
    const size_t tk = MT * (size_t)KVH_ * 32;
    rope_kernel<<<(unsigned)((tk + 255) / 256), 256, 0, stream>>>(kb, freq, KVH_, tk);
  }

  attn_kernel<<<dim3(B_ * H_, T_ / 64), dim3(128), 0, stream>>>(qb, kb, vb, ab);

  gemm_wmma_kernel<__bf16, float, float>
      <<<dim3(C_ / 128, (int)(MT / 128)), blk, 0, stream>>>(
          ab, Wo, (float*)d_out, (int)MT, C_, C_);
}
